// MemoryWrapLayer_27642409517342
// MI455X (gfx1250) — matrix operand order repaired and checked by reading, not compiled, not run on hardware
//
#include <hip/hip_runtime.h>
#include <hip/hip_bf16.h>
#include <stdint.h>

// ---------------------------------------------------------------------------
// MemoryWrapLayer on MI455X (gfx1250): bf16 WMMA GEMM pipeline with
// double-buffered GLOBAL_LOAD_ASYNC_TO_LDS staging (ASYNCcnt-tracked).
//   1) normalize+cast enc  -> enc_n (bf16), finalin[:, :D] = enc (bf16)
//   2) normalize+cast mem  -> mem_n (bf16), memT (bf16, transposed raw mem)
//   3) cast fc1_w / fc2_w  -> bf16 (fc2 padded 1000->1024 rows)
//   4) GEMM sim   = enc_n @ mem_n^T                      (f32 out)
//   5) sparsemax rows of sim (binary-search tau)          -> cw (bf16)
//   6) GEMM mv    = cw @ mem  (Bt = memT)  -> finalin[:, D:2D] (bf16)
//   7) GEMM hid   = relu(finalin @ fc1_w^T + b1)          -> hidden (bf16)
//   8) GEMM out   = hidden @ fc2_w^T + b2                 -> d_out (f32)
// ---------------------------------------------------------------------------

typedef unsigned short u16;
typedef __attribute__((ext_vector_type(16))) __bf16 v16bf;
typedef __attribute__((ext_vector_type(8)))  float  v8f;

#define B_SZ    4096
#define M_SZ    8192
#define DIM_SZ  2048
#define OUT_SZ  1000
#define OUTP_SZ 1024          // fc2 N padded to tile multiple
#define HID_SZ  8192

#define BM   128
#define BN   128
#define BK   32
#define LDSS 40               // padded LDS row stride in halves (80B: bank-conflict-free)

struct __align__(16) U4 { uint32_t x, y, z, w; };
union Frag { v16bf v; U4 q[2]; };

__device__ __forceinline__ u16 f2bf(float f) {
  uint32_t u = __float_as_uint(f);
  u += 0x7FFFu + ((u >> 16) & 1u);   // round-to-nearest-even
  return (u16)(u >> 16);
}

// --- CDNA5 async copy: global -> LDS, 16B per lane, tracked by ASYNCcnt ----
// ISA form: GLOBAL_LOAD_ASYNC_TO_LDS_B128  VDST(=LDS byte addr), VADDR, off
__device__ __forceinline__ void async_b128(uint32_t lds_off, const u16* g) {
  asm volatile("global_load_async_to_lds_b128 %0, %1, off"
               :: "v"(lds_off), "v"(g) : "memory");
}

__device__ __forceinline__ void wait_async0() {
  asm volatile("s_wait_asynccnt 0x0" ::: "memory");
}

__device__ __forceinline__ uint32_t lds_off32(const void* p) {
  // generic (flat) shared pointer: ADDR[31:0] is the LDS byte address
  return (uint32_t)(uintptr_t)p;
}

// ---------------------------------------------------------------------------
// Generic bf16 GEMM: C[M,N] = A[M,K] * Bt[N,K]^T  (+bias, relu), f32 accum.
// 256 threads (8 waves). Block tile 128x128, wave tile 32x64 (2x4 WMMA),
// K step 32, double-buffered async LDS staging.
// ---------------------------------------------------------------------------
__global__ __launch_bounds__(256)
void gemm_bf16_kernel(const u16* __restrict__ A, const u16* __restrict__ Bt,
                      const float* __restrict__ bias,
                      float* __restrict__ outF, u16* __restrict__ outB,
                      int N, int Nreal, int K, int ldc, int relu)
{
  __shared__ __align__(16) u16 As[2][BM * LDSS];
  __shared__ __align__(16) u16 Bs[2][BN * LDSS];

  const int tid  = threadIdx.x;
  const int bm   = blockIdx.y * BM;
  const int bn   = blockIdx.x * BN;
  const int w    = tid >> 5;
  const int lane = tid & 31;
  const int wm   = (w >> 1) * 32;          // 4 waves along M (32 rows each)
  const int wn   = (w & 1) * 64;           // 2 waves along N (64 cols each)
  const int half = lane >> 4;
  const int r    = lane & 15;

  // per-thread staging assignment: rows (tid>>2) and (tid>>2)+64, chunk (tid&3)
  const int srow = tid >> 2;               // 0..63
  const int scc  = (tid & 3) * 8;          // half offset within 32-half row
  const u16* gA0 = A  + (size_t)(bm + srow)      * K + scc;
  const u16* gA1 = A  + (size_t)(bm + srow + 64) * K + scc;
  const u16* gB0 = Bt + (size_t)(bn + srow)      * K + scc;
  const u16* gB1 = Bt + (size_t)(bn + srow + 64) * K + scc;

  uint32_t lA0[2], lA1[2], lB0[2], lB1[2];
#pragma unroll
  for (int b = 0; b < 2; ++b) {
    lA0[b] = lds_off32(&As[b][srow * LDSS + scc]);
    lA1[b] = lds_off32(&As[b][(srow + 64) * LDSS + scc]);
    lB0[b] = lds_off32(&Bs[b][srow * LDSS + scc]);
    lB1[b] = lds_off32(&Bs[b][(srow + 64) * LDSS + scc]);
  }

  v8f acc[2][4] = {};

  // prologue: stage tile 0 into buffer 0
  async_b128(lA0[0], gA0);
  async_b128(lA1[0], gA1);
  async_b128(lB0[0], gB0);
  async_b128(lB1[0], gB1);

  int cur = 0;
  for (int k0 = 0; k0 < K; k0 += BK) {
    wait_async0();            // this wave's async writes for tile `cur` done
    __syncthreads();          // ... and everyone else's too

    if (k0 + BK < K) {        // stage next tile into the other buffer
      int nb = cur ^ 1;
      async_b128(lA0[nb], gA0 + k0 + BK);
      async_b128(lA1[nb], gA1 + k0 + BK);
      async_b128(lB0[nb], gB0 + k0 + BK);
      async_b128(lB1[nb], gB1 + k0 + BK);
    }

    // ---- fragment fetch per documented VGPR layouts ----
    Frag a[2], b[4];
#pragma unroll
    for (int mi = 0; mi < 2; ++mi) {       // A 16x32: k chunks at half*8, 16+half*8
      int arow = wm + mi * 16 + r;
      a[mi].q[0] = *(const U4*)&As[cur][arow * LDSS + half * 8];
      a[mi].q[1] = *(const U4*)&As[cur][arow * LDSS + 16 + half * 8];
    }
#pragma unroll
    for (int ni = 0; ni < 4; ++ni) {       // B 32x16: k chunks at half*8, 16+half*8
      int brow = wn + ni * 16 + r;
      b[ni].q[0] = *(const U4*)&Bs[cur][brow * LDSS + half * 8];
      b[ni].q[1] = *(const U4*)&Bs[cur][brow * LDSS + 16 + half * 8];
    }

#pragma unroll
    for (int mi = 0; mi < 2; ++mi)
#pragma unroll
      for (int ni = 0; ni < 4; ++ni)
        acc[mi][ni] = __builtin_amdgcn_wmma_f32_16x16x32_bf16(
            false, a[mi].v, false, b[ni].v, (short)0, acc[mi][ni], false, false);

    __syncthreads();          // reads of buffer `cur` done before it is reused
    cur ^= 1;
  }

  // ---- epilogue: C lane mapping m = e + 8*half, n = lane&15 ----
#pragma unroll
  for (int mi = 0; mi < 2; ++mi)
#pragma unroll
    for (int ni = 0; ni < 4; ++ni) {
      int n = bn + wn + ni * 16 + r;
      if (n < Nreal) {
        float bv = bias ? bias[n] : 0.0f;
#pragma unroll
        for (int e = 0; e < 8; ++e) {
          int m = bm + wm + mi * 16 + e + 8 * half;
          float vv = acc[mi][ni][e] + bv;
          if (relu) vv = fmaxf(vv, 0.0f);
          if (outF) outF[(size_t)m * ldc + n] = vv;
          if (outB) outB[(size_t)m * ldc + n] = f2bf(vv);
        }
      }
    }
}

// ---------------------------------------------------------------------------
// Row-normalize encoder rows; also drop raw enc (bf16) into finalin[:, :D].
// ---------------------------------------------------------------------------
__global__ __launch_bounds__(256)
void norm_enc_kernel(const float* __restrict__ enc, u16* __restrict__ enc_n,
                     u16* __restrict__ finalin)
{
  __shared__ float red[256];
  const int row = blockIdx.x, tid = threadIdx.x;
  const float* er = enc + (size_t)row * DIM_SZ;
  float s = 0.f;
  for (int d = tid; d < DIM_SZ; d += 256) { float v = er[d]; s += v * v; }
  red[tid] = s; __syncthreads();
  for (int st = 128; st > 0; st >>= 1) { if (tid < st) red[tid] += red[tid + st]; __syncthreads(); }
  const float inv = 1.0f / sqrtf(red[0] + 1e-6f);
  for (int d = tid; d < DIM_SZ; d += 256) {
    float v = er[d];
    enc_n[(size_t)row * DIM_SZ + d]        = f2bf(v * inv);
    finalin[(size_t)row * 2 * DIM_SZ + d]  = f2bf(v);
  }
}

// ---------------------------------------------------------------------------
// Row-normalize memory rows (-> mem_n bf16) and emit raw mem^T (bf16).
// ---------------------------------------------------------------------------
__global__ __launch_bounds__(256)
void norm_mem_kernel(const float* __restrict__ mem, u16* __restrict__ mem_n,
                     u16* __restrict__ memT)
{
  __shared__ float red[256];
  const int row = blockIdx.x, tid = threadIdx.x;
  const float* mr = mem + (size_t)row * DIM_SZ;
  float s = 0.f;
  for (int d = tid; d < DIM_SZ; d += 256) { float v = mr[d]; s += v * v; }
  red[tid] = s; __syncthreads();
  for (int st = 128; st > 0; st >>= 1) { if (tid < st) red[tid] += red[tid + st]; __syncthreads(); }
  const float inv = 1.0f / sqrtf(red[0] + 1e-6f);
  for (int d = tid; d < DIM_SZ; d += 256) {
    float v = mr[d];
    mem_n[(size_t)row * DIM_SZ + d] = f2bf(v * inv);
    memT[(size_t)d * M_SZ + row]    = f2bf(v);
  }
}

// ---------------------------------------------------------------------------
// f32 -> bf16 cast with zero-pad tail (fc2 rows 1000..1023 zeroed).
// ---------------------------------------------------------------------------
__global__ __launch_bounds__(256)
void cast_pad_kernel(const float* __restrict__ src, u16* __restrict__ dst,
                     size_t nsrc, size_t ndst)
{
  size_t stride = (size_t)gridDim.x * blockDim.x;
  for (size_t i = (size_t)blockIdx.x * blockDim.x + threadIdx.x; i < ndst; i += stride)
    dst[i] = (i < nsrc) ? f2bf(src[i]) : (u16)0;
}

// ---------------------------------------------------------------------------
// Sparsemax per row (length M_SZ), sort-free: binary-search tau on the
// monotone function f(tau)=sum(max(z-tau,0)), then exact tau from support.
// ---------------------------------------------------------------------------
__global__ __launch_bounds__(256)
void sparsemax_kernel(const float* __restrict__ sim, u16* __restrict__ cw)
{
  __shared__ float z[M_SZ];
  __shared__ float red[256];
  const int row = blockIdx.x, tid = threadIdx.x;
  const float* zr = sim + (size_t)row * M_SZ;

  float mx = -1e30f;
  for (int j = tid; j < M_SZ; j += 256) { float v = zr[j]; z[j] = v; mx = fmaxf(mx, v); }
  red[tid] = mx; __syncthreads();
  for (int st = 128; st > 0; st >>= 1) { if (tid < st) red[tid] = fmaxf(red[tid], red[tid + st]); __syncthreads(); }
  mx = red[0]; __syncthreads();

  float lo = mx - 1.0f, hi = mx;            // f(mx-1)>=1 >= f(mx)=0
  for (int it = 0; it < 26; ++it) {
    float mid = 0.5f * (lo + hi);
    float s = 0.f;
    for (int j = tid; j < M_SZ; j += 256) s += fmaxf(z[j] - mid, 0.f);
    red[tid] = s; __syncthreads();
    for (int st = 128; st > 0; st >>= 1) { if (tid < st) red[tid] += red[tid + st]; __syncthreads(); }
    s = red[0]; __syncthreads();
    if (s > 1.0f) lo = mid; else hi = mid;
  }

  const float tau0 = 0.5f * (lo + hi);
  float cnt = 0.f, sum = 0.f;
  for (int j = tid; j < M_SZ; j += 256)
    if (z[j] > tau0) { cnt += 1.f; sum += z[j]; }
  red[tid] = cnt; __syncthreads();
  for (int st = 128; st > 0; st >>= 1) { if (tid < st) red[tid] += red[tid + st]; __syncthreads(); }
  cnt = red[0]; __syncthreads();
  red[tid] = sum; __syncthreads();
  for (int st = 128; st > 0; st >>= 1) { if (tid < st) red[tid] += red[tid + st]; __syncthreads(); }
  sum = red[0]; __syncthreads();

  const float tau = (sum - 1.0f) / fmaxf(cnt, 1.0f);
  for (int j = tid; j < M_SZ; j += 256)
    cw[(size_t)row * M_SZ + j] = f2bf(fmaxf(z[j] - tau, 0.f));
}

// ---------------------------------------------------------------------------
extern "C" void kernel_launch(void* const* d_in, const int* in_sizes, int n_in,
                              void* d_out, int out_size, void* d_ws, size_t ws_size,
                              hipStream_t stream)
{
  const float* enc  = (const float*)d_in[0];
  const float* mem  = (const float*)d_in[1];
  const float* fc1w = (const float*)d_in[2];
  const float* fc1b = (const float*)d_in[3];
  const float* fc2w = (const float*)d_in[4];
  const float* fc2b = (const float*)d_in[5];

  char* ws = (char*)d_ws;
  size_t off = 0;
  auto take = [&](size_t bytes) -> char* {
    char* p = ws + off; off += (bytes + 255) & ~(size_t)255; return p;
  };
  u16*  enc_n   = (u16*)take((size_t)B_SZ * DIM_SZ * 2);
  u16*  mem_n   = (u16*)take((size_t)M_SZ * DIM_SZ * 2);
  u16*  memT    = (u16*)take((size_t)DIM_SZ * M_SZ * 2);
  u16*  cw      = (u16*)take((size_t)B_SZ * M_SZ * 2);
  u16*  finalin = (u16*)take((size_t)B_SZ * 2 * DIM_SZ * 2);
  u16*  fc1wb   = (u16*)take((size_t)HID_SZ * 2 * DIM_SZ * 2);
  u16*  fc2wb   = (u16*)take((size_t)OUTP_SZ * HID_SZ * 2);
  float* sim    = (float*)take((size_t)B_SZ * M_SZ * 4);
  u16*  hidden  = (u16*)sim;   // alias: sim is dead after sparsemax

  // 1-3: precision prep
  cast_pad_kernel<<<2048, 256, 0, stream>>>(fc1w, fc1wb,
      (size_t)HID_SZ * 2 * DIM_SZ, (size_t)HID_SZ * 2 * DIM_SZ);
  cast_pad_kernel<<<2048, 256, 0, stream>>>(fc2w, fc2wb,
      (size_t)OUT_SZ * HID_SZ, (size_t)OUTP_SZ * HID_SZ);
  norm_enc_kernel<<<B_SZ, 256, 0, stream>>>(enc, enc_n, finalin);
  norm_mem_kernel<<<M_SZ, 256, 0, stream>>>(mem, mem_n, memT);

  // 4: sim = enc_n @ mem_n^T  -> f32 [B, M]
  gemm_bf16_kernel<<<dim3(M_SZ / BN, B_SZ / BM), 256, 0, stream>>>(
      enc_n, mem_n, nullptr, sim, nullptr, M_SZ, M_SZ, DIM_SZ, M_SZ, 0);

  // 5: sparsemax rows -> cw bf16
  sparsemax_kernel<<<B_SZ, 256, 0, stream>>>(sim, cw);

  // 6: memory_vector = cw @ mem -> finalin[:, D:2D] (bf16)
  gemm_bf16_kernel<<<dim3(DIM_SZ / BN, B_SZ / BM), 256, 0, stream>>>(
      cw, memT, nullptr, nullptr, finalin + DIM_SZ, DIM_SZ, DIM_SZ, M_SZ,
      2 * DIM_SZ, 0);

  // 7: hidden = relu(finalin @ fc1_w^T + b1) -> bf16
  gemm_bf16_kernel<<<dim3(HID_SZ / BN, B_SZ / BM), 256, 0, stream>>>(
      finalin, fc1wb, fc1b, nullptr, hidden, HID_SZ, HID_SZ, 2 * DIM_SZ,
      HID_SZ, 1);

  // 8: out = hidden @ fc2_w^T + b2 -> f32 d_out [B, 1000]
  gemm_bf16_kernel<<<dim3(OUTP_SZ / BN, B_SZ / BM), 256, 0, stream>>>(
      hidden, fc2wb, fc2b, (float*)d_out, nullptr, OUTP_SZ, OUT_SZ, HID_SZ,
      OUT_SZ, 0);
}
